// Attention_51556787421487
// MI455X (gfx1250) — compile-verified, matrix operand order checked by reading, not run on hardware
//
#include <hip/hip_runtime.h>
#include <hip/hip_bf16.h>

typedef __attribute__((ext_vector_type(16))) _Float16 v16h;
typedef __attribute__((ext_vector_type(8)))  _Float16 v8h;
typedef __attribute__((ext_vector_type(8)))  float    v8f;
typedef __attribute__((ext_vector_type(4)))  float    v4f;
typedef __attribute__((ext_vector_type(4)))  int      v4i;

union U16 { v16h v; v8h h[2]; };

#define AS1 __attribute__((address_space(1)))
#define AS3 __attribute__((address_space(3)))

#if defined(__has_builtin)
#if __has_builtin(__builtin_amdgcn_global_load_async_to_lds_b128)
#define HAVE_ASYNC_LDS 1
#endif
#endif
#ifndef HAVE_ASYNC_LDS
#define HAVE_ASYNC_LDS 0
#endif

__device__ __forceinline__ v8f wmma_f16(v16h a, v16h b, v8f c) {
    return __builtin_amdgcn_wmma_f32_16x16x32_f16(false, a, false, b, (short)0, c,
                                                  false, false);
}

// Copy 16B global -> LDS. Async (ASYNCcnt-tracked) when available.
__device__ __forceinline__ void copy16_g2l(const _Float16* g, _Float16* l) {
#if HAVE_ASYNC_LDS
    __builtin_amdgcn_global_load_async_to_lds_b128((AS1 v4i*)g, (AS3 v4i*)l, 0, 0);
#else
    *(v8h*)l = *(const v8h*)g;
#endif
}

__device__ __forceinline__ void wait_async_copies() {
#if HAVE_ASYNC_LDS
#if __has_builtin(__builtin_amdgcn_s_wait_asynccnt)
    __builtin_amdgcn_s_wait_asynccnt(0);
#else
    asm volatile("s_wait_asynccnt 0x0" ::: "memory");
#endif
#endif
}

constexpr int NWIN  = 4096;      // 16*16*16 windows
constexpr int NTOK  = 49;        // 7*7 tokens per window
constexpr int DIM   = 256;
constexpr int HEADS = 8;
constexpr int MTOT  = NWIN * NTOK;   // 200704 rows
constexpr int NQKV  = 3 * DIM;       // 768
constexpr int MBLKS = MTOT / 64;     // 3136
constexpr int LDSP  = 72;            // padded LDS row stride (halfs): 144B, 16B-aligned

// ---------------------------------------------------------------------------
// Kernel 1: convert weights to f16, transposed to [N][K] for B-fragment loads
// ---------------------------------------------------------------------------
__global__ void cvt_weights(const float* __restrict__ wqkv,
                            const float* __restrict__ wout,
                            _Float16* __restrict__ wq_t,
                            _Float16* __restrict__ wo_t) {
    int i = blockIdx.x * 256 + threadIdx.x;
    if (i < DIM * NQKV) {                 // wqkv: [K=256][N=768] -> wq_t[N][K]
        int k = i / NQKV, n = i % NQKV;
        wq_t[n * DIM + k] = (_Float16)wqkv[i];
    }
    if (i < DIM * DIM) {                  // wout: [K=256][N=256] -> wo_t[N][K]
        int k = i / DIM, n = i % DIM;
        wo_t[n * DIM + k] = (_Float16)wout[i];
    }
}

// ---------------------------------------------------------------------------
// Kernel 2: QKV GEMM  (200704 x 256) * (256 x 768) -> f16 qkv
// Block tile 64(M) x 128(N), 8 waves. A (f32->f16 conversion) staged through
// VGPRs into fragment-order LDS; B (already f16) staged with async global->LDS.
// ---------------------------------------------------------------------------
__global__ __launch_bounds__(256)
void qkv_gemm(const float* __restrict__ X, const _Float16* __restrict__ Wt,
              _Float16* __restrict__ Y) {
    __shared__ _Float16 lA[4 * 32 * 16];   // 64x32 A tile, fragment order
    __shared__ _Float16 lB[128 * LDSP];    // 128x32 B tile, natural rows (padded)

    const int tid  = threadIdx.x;
    const int lane = tid & 31;
    const int wid  = tid >> 5;
    const int tl   = lane & 15;
    const int kb   = 8 * (lane >> 4);
    const int mblk = blockIdx.x % MBLKS;
    const int nblk = blockIdx.x / MBLKS;
    const int m0 = mblk * 64;
    const int n0 = nblk * 128;
    const int wm = wid & 3;     // m-tile of this wave
    const int wn = wid >> 2;    // n half (covers 4 n-tiles = 64 cols)

    v8f acc[4] = {};

    for (int k0 = 0; k0 < DIM; k0 += 32) {
        if (k0 + 32 < DIM)   // prefetch next A tile -> global_prefetch_b8
            __builtin_prefetch(X + (size_t)(m0 + (tid >> 3)) * DIM + k0 + 32, 0, 0);
        // B: 128x32 f16 tile, async global->LDS (2 x 16B per thread)
#pragma unroll
        for (int it = 0; it < 2; ++it) {
            int idx = tid + it * 256;           // 0..511
            int n   = idx >> 2;                 // out-col 0..127
            int c8  = (idx & 3) * 8;            // k 0,8,16,24
            copy16_g2l(Wt + (size_t)(n0 + n) * DIM + k0 + c8, &lB[n * LDSP + c8]);
        }
        // A: 64x32 f32 -> f16 fragment-order LDS; 2 x v4f per thread (NT loads)
#pragma unroll
        for (int it = 0; it < 2; ++it) {
            int idx = tid + it * 256;           // 0..511
            int r   = idx >> 3;                 // row 0..63
            int c4  = (idx & 7) * 4;            // k 0..28
            const v4f f =
                __builtin_nontemporal_load((const v4f*)(X + (size_t)(m0 + r) * DIM + k0 + c4));
            int hi  = (c4 >> 3) & 1;
            int l   = (r & 15) + 16 * hi;
            int mt  = r >> 4;
            int pos = (c4 & 7) + ((c4 >> 4) << 3);
            _Float16* dst = &lA[(mt * 32 + l) * 16 + pos];
            dst[0] = (_Float16)f[0]; dst[1] = (_Float16)f[1];
            dst[2] = (_Float16)f[2]; dst[3] = (_Float16)f[3];
        }
        wait_async_copies();
        __syncthreads();
        v16h a = *(const v16h*)&lA[(wm * 32 + lane) * 16];
#pragma unroll
        for (int j = 0; j < 4; ++j) {
            int nt = wn * 4 + j;
            U16 bf;
            const _Float16* bp = &lB[(nt * 16 + tl) * LDSP + kb];
            bf.h[0] = *(const v8h*)bp;
            bf.h[1] = *(const v8h*)(bp + 16);
            acc[j] = wmma_f16(a, bf.v, acc[j]);
        }
        __syncthreads();
    }
    // Store C (16x16 f32 layout: row = v + 8*(lane>>4), col = lane&15)
    const int rlo = 8 * (lane >> 4);
    const int cc  = lane & 15;
#pragma unroll
    for (int j = 0; j < 4; ++j) {
        int col = n0 + wn * 64 + j * 16 + cc;
#pragma unroll
        for (int v = 0; v < 8; ++v) {
            int row = m0 + wm * 16 + v + rlo;
            __builtin_nontemporal_store((_Float16)acc[j][v], Y + (size_t)row * NQKV + col);
        }
    }
}

// ---------------------------------------------------------------------------
// Kernel 3: attention per (window, head). 4 waves, each owns a 16-row tile of
// the 64-padded 49x49 score matrix. QK^T and P*V via v_wmma_f32_16x16x32_f16.
// ---------------------------------------------------------------------------
__global__ __launch_bounds__(128)
void attn_kernel(const _Float16* __restrict__ QKV,
                 const float* __restrict__ bias_table,
                 _Float16* __restrict__ Att) {
    const int w    = blockIdx.x;
    const int h    = blockIdx.y;
    const int tid  = threadIdx.x;
    const int lane = tid & 31;
    const int wid  = tid >> 5;          // m-tile 0..3
    const float SCALE = 0.0625f;        // 256^-0.5

    __shared__ float    lbias[NTOK * NTOK];   // per-head relative bias tile
    __shared__ _Float16 lP[64 * LDSP];        // softmaxed probs (padded rows)
    __shared__ _Float16 lVt[32 * LDSP];       // V transposed [dh][token]

    // Relative-position bias tile for this head
    for (int i = tid; i < NTOK * NTOK; i += 128) {
        int t = i / NTOK, u = i % NTOK;
        int idx = (t / 7 - u / 7 + 6) * 13 + (t % 7 - u % 7 + 6);
        lbias[i] = bias_table[idx * HEADS + h];
    }
    // V transposed into LDS, zero-padded tokens (WMMA: NaN in -> NaN out)
    for (int i = tid; i < NTOK * 32; i += 128) {
        int u = i >> 5, c = i & 31;
        lVt[c * LDSP + u] = QKV[(size_t)(w * NTOK + u) * NQKV + 512 + h * 32 + c];
    }
    for (int i = tid; i < 32 * 23; i += 128) {
        int c = i / 23, u = NTOK + i % 23;
        lVt[c * LDSP + u] = (_Float16)0.f;
    }

    // Q fragment (rows 16*wid..): lane covers row t = 16*wid + lane%16
    const int tl = lane & 15;
    const int kb = 8 * (lane >> 4);
    U16 qf; qf.v = (v16h)(_Float16)0.f;
    {
        int t = wid * 16 + tl;
        if (t < NTOK) {
            const _Float16* p = QKV + (size_t)(w * NTOK + t) * NQKV + h * 32 + kb;
            qf.h[0] = *(const v8h*)p;
            qf.h[1] = *(const v8h*)(p + 16);
        }
    }
    // K fragments per n-tile (column tokens)
    U16 kf[4];
    v8f s[4] = {};
#pragma unroll
    for (int nt = 0; nt < 4; ++nt) {
        kf[nt].v = (v16h)(_Float16)0.f;
        int u = nt * 16 + tl;
        if (u < NTOK) {
            const _Float16* p = QKV + (size_t)(w * NTOK + u) * NQKV + 256 + h * 32 + kb;
            kf[nt].h[0] = *(const v8h*)p;
            kf[nt].h[1] = *(const v8h*)(p + 16);
        }
        s[nt] = wmma_f16(qf.v, kf[nt].v, s[nt]);
    }
    __syncthreads();   // lbias / lVt ready

    // scale + bias + mask + row softmax (row spread over 16-lane group + 4 accs)
    const int rlo = 8 * (lane >> 4);
    const int cc  = lane & 15;
    float sv[4][8];
#pragma unroll
    for (int v = 0; v < 8; ++v) {
        int t = wid * 16 + v + rlo;
#pragma unroll
        for (int nt = 0; nt < 4; ++nt) {
            int u = nt * 16 + cc;
            float x = s[nt][v] * SCALE;
            if (t < NTOK && u < NTOK) x += lbias[t * NTOK + u];
            if (u >= NTOK) x = -1e30f;
            sv[nt][v] = x;
        }
        float m = fmaxf(fmaxf(sv[0][v], sv[1][v]), fmaxf(sv[2][v], sv[3][v]));
        m = fmaxf(m, __shfl_xor(m, 1, 32));
        m = fmaxf(m, __shfl_xor(m, 2, 32));
        m = fmaxf(m, __shfl_xor(m, 4, 32));
        m = fmaxf(m, __shfl_xor(m, 8, 32));
        float sum = 0.f;
#pragma unroll
        for (int nt = 0; nt < 4; ++nt) {
            float e = __expf(sv[nt][v] - m);
            sv[nt][v] = e;
            sum += e;
        }
        sum += __shfl_xor(sum, 1, 32);
        sum += __shfl_xor(sum, 2, 32);
        sum += __shfl_xor(sum, 4, 32);
        sum += __shfl_xor(sum, 8, 32);
        float inv = __frcp_rn(sum);
        int row = wid * 16 + v + rlo;
#pragma unroll
        for (int nt = 0; nt < 4; ++nt)
            lP[row * LDSP + nt * 16 + cc] = (_Float16)(sv[nt][v] * inv);
    }
    __syncthreads();   // lP complete

    // P (A-fragment, K=64 -> 2 k-steps) x V (B-fragment from lVt)
    U16 pa[2], vb[2][2];
#pragma unroll
    for (int ks = 0; ks < 2; ++ks) {
        const _Float16* p = &lP[(wid * 16 + tl) * LDSP + ks * 32 + kb];
        pa[ks].h[0] = *(const v8h*)p;
        pa[ks].h[1] = *(const v8h*)(p + 16);
#pragma unroll
        for (int nt = 0; nt < 2; ++nt) {
            const _Float16* q = &lVt[(nt * 16 + tl) * LDSP + ks * 32 + kb];
            vb[nt][ks].h[0] = *(const v8h*)q;
            vb[nt][ks].h[1] = *(const v8h*)(q + 16);
        }
    }
    v8f o[2] = {};
#pragma unroll
    for (int nt = 0; nt < 2; ++nt)
#pragma unroll
        for (int ks = 0; ks < 2; ++ks)
            o[nt] = wmma_f16(pa[ks].v, vb[nt][ks].v, o[nt]);

    // Store head output (f16) into [m][256] attention buffer
#pragma unroll
    for (int nt = 0; nt < 2; ++nt) {
#pragma unroll
        for (int v = 0; v < 8; ++v) {
            int t = wid * 16 + v + rlo;
            if (t < NTOK)
                Att[(size_t)(w * NTOK + t) * DIM + h * 32 + nt * 16 + cc] =
                    (_Float16)o[nt][v];
        }
    }
}

// ---------------------------------------------------------------------------
// Kernel 4: output projection  (200704 x 256) f16 * (256 x 256) f16 -> f32
// Both A and B tiles staged with async global->LDS copies.
// ---------------------------------------------------------------------------
__global__ __launch_bounds__(256)
void out_gemm(const _Float16* __restrict__ A, const _Float16* __restrict__ Wt,
              float* __restrict__ Y) {
    __shared__ _Float16 lA[64 * LDSP];
    __shared__ _Float16 lB[128 * LDSP];

    const int tid  = threadIdx.x;
    const int lane = tid & 31;
    const int wid  = tid >> 5;
    const int tl   = lane & 15;
    const int kb   = 8 * (lane >> 4);
    const int mblk = blockIdx.x % MBLKS;
    const int nblk = blockIdx.x / MBLKS;
    const int m0 = mblk * 64;
    const int n0 = nblk * 128;
    const int wm = wid & 3;
    const int wn = wid >> 2;

    v8f acc[4] = {};

    for (int k0 = 0; k0 < DIM; k0 += 32) {
        // A: 64x32 halfs, async; one 16B chunk per thread (256 chunks)
        {
            int r  = tid >> 2;
            int c8 = (tid & 3) * 8;
            copy16_g2l(A + (size_t)(m0 + r) * DIM + k0 + c8, &lA[r * LDSP + c8]);
        }
        // B: 128x32 halfs, async; 2 x 16B per thread
#pragma unroll
        for (int it = 0; it < 2; ++it) {
            int idx = tid + it * 256;
            int n   = idx >> 2;
            int c8  = (idx & 3) * 8;
            copy16_g2l(Wt + (size_t)(n0 + n) * DIM + k0 + c8, &lB[n * LDSP + c8]);
        }
        wait_async_copies();
        __syncthreads();
        U16 af;
        const _Float16* ap = &lA[(wm * 16 + tl) * LDSP + kb];
        af.h[0] = *(const v8h*)ap;
        af.h[1] = *(const v8h*)(ap + 16);
#pragma unroll
        for (int j = 0; j < 4; ++j) {
            int nt = wn * 4 + j;
            U16 bf;
            const _Float16* bp = &lB[(nt * 16 + tl) * LDSP + kb];
            bf.h[0] = *(const v8h*)bp;
            bf.h[1] = *(const v8h*)(bp + 16);
            acc[j] = wmma_f16(af.v, bf.v, acc[j]);
        }
        __syncthreads();
    }
    const int rlo = 8 * (lane >> 4);
    const int cc  = lane & 15;
#pragma unroll
    for (int j = 0; j < 4; ++j) {
        int col = n0 + wn * 64 + j * 16 + cc;
#pragma unroll
        for (int v = 0; v < 8; ++v) {
            int row = m0 + wm * 16 + v + rlo;
            __builtin_nontemporal_store(acc[j][v], Y + (size_t)row * DIM + col);
        }
    }
}

// ---------------------------------------------------------------------------
extern "C" void kernel_launch(void* const* d_in, const int* in_sizes, int n_in,
                              void* d_out, int out_size, void* d_ws, size_t ws_size,
                              hipStream_t stream) {
    const float* x      = (const float*)d_in[0];
    const float* w_qkv  = (const float*)d_in[1];
    const float* w_out  = (const float*)d_in[2];
    const float* bias_t = (const float*)d_in[3];
    float* out = (float*)d_out;

    char* ws = (char*)d_ws;
    _Float16* wq_t = (_Float16*)(ws);                                  // 768*256 f16
    _Float16* wo_t = (_Float16*)(ws + (size_t)NQKV * DIM * 2);         // 256*256 f16
    char* p = ws + (size_t)NQKV * DIM * 2 + (size_t)DIM * DIM * 2;
    _Float16* qkv = (_Float16*)p;                                      // M*768 f16
    _Float16* att = (_Float16*)(p + (size_t)MTOT * NQKV * 2);          // M*256 f16

    cvt_weights<<<(DIM * NQKV + 255) / 256, 256, 0, stream>>>(w_qkv, w_out, wq_t, wo_t);
    qkv_gemm<<<MBLKS * (NQKV / 128), 256, 0, stream>>>(x, wq_t, qkv);
    attn_kernel<<<dim3(NWIN, HEADS), 128, 0, stream>>>(qkv, bias_t, att);
    out_gemm<<<MBLKS * (DIM / 128), 256, 0, stream>>>(att, wo_t, out);
}
